// MultiHeadAttentionUnit_87505663688897
// MI455X (gfx1250) — matrix operand order repaired and checked by reading, not compiled, not run on hardware
//
#include <hip/hip_runtime.h>

// ---------------------------------------------------------------------------
// Problem constants (from reference): B=8, S=1024, D=128, H=16, E=2048
// ---------------------------------------------------------------------------
#define BB   8
#define SS   1024
#define DD   128
#define HH   16
#define HD   2048   // H*D
#define EE   2048

typedef __attribute__((ext_vector_type(16))) __bf16 v16bf;
typedef __attribute__((ext_vector_type(8)))  __bf16 v8bf;
typedef __attribute__((ext_vector_type(8)))  float  v8f;
typedef __attribute__((ext_vector_type(4)))  int    v4i;

union AFragU { v16bf v; v8bf h[2]; };

// ---------------------------------------------------------------------------
// CDNA5 async global->LDS copy (ASYNCcnt path), with synchronous fallback.
// Builtin signature (from hipcc diagnostic): param0 = int4 addrspace(1)*,
// param1 = int4 addrspace(3)*, then imm offset, imm cpol.
// ---------------------------------------------------------------------------
#if defined(__has_builtin)
#  if __has_builtin(__builtin_amdgcn_global_load_async_to_lds_b128)
#    define HAVE_ASYNC_LDS 1
#  endif
#endif
#ifndef HAVE_ASYNC_LDS
#  define HAVE_ASYNC_LDS 0
#endif

static __device__ __forceinline__ void cp_async_b128(void* lds, const void* g) {
#if HAVE_ASYNC_LDS
    __builtin_amdgcn_global_load_async_to_lds_b128(
        (__attribute__((address_space(1))) v4i*)g,
        (__attribute__((address_space(3))) v4i*)lds, 0, 0);
#else
    *(uint4*)lds = *(const uint4*)g;
#endif
}

static __device__ __forceinline__ void async_wait0() {
#if HAVE_ASYNC_LDS
#  if __has_builtin(__builtin_amdgcn_s_wait_asynccnt)
    __builtin_amdgcn_s_wait_asynccnt(0);
#  else
    asm volatile("s_wait_asynccnt 0" ::: "memory");
#  endif
#endif
}

static __device__ __forceinline__ unsigned short f32_to_bf16(float f) {
    unsigned int u = __float_as_uint(f);
    unsigned int r = u + 0x7FFFu + ((u >> 16) & 1u);   // round-to-nearest-even
    return (unsigned short)(r >> 16);
}

static __device__ __forceinline__ v8f vzero8() {
    v8f z;
#pragma unroll
    for (int i = 0; i < 8; ++i) z[i] = 0.0f;
    return z;
}

// A-fragment (16x32 bf16): lane holds row M = lane&15.
// elements 0..7  = K {hi*8 .. hi*8+7},  elements 8..15 = K {16+hi*8 ..}
static __device__ __forceinline__ v16bf frag_a(const unsigned short* rowptr, int hi) {
    AFragU f;
    f.h[0] = *(const v8bf*)(rowptr + hi * 8);
    f.h[1] = *(const v8bf*)(rowptr + 16 + hi * 8);
    return f.v;
}

// B-fragment (32x16 bf16) from B^T rows (32 contiguous K per N-row):
// lane holds col N = lane&15, elements 0..7 = K {hi*8 ..}, 8..15 = K {16+hi*8 ..}
static __device__ __forceinline__ v16bf frag_b(const unsigned short* rowptr, int hi) {
    AFragU f;
    f.h[0] = *(const v8bf*)(rowptr + hi * 8);
    f.h[1] = *(const v8bf*)(rowptr + 16 + hi * 8);
    return f.v;
}

static __device__ __forceinline__ v8f wmma_bf16(v16bf a, v16bf b, v8f c) {
    return __builtin_amdgcn_wmma_f32_16x16x32_bf16(false, a, false, b,
                                                   (short)0, c, false, false);
}

// ---------------------------------------------------------------------------
// Kernel 1: transpose fp32 weight [rows][cols] -> bf16 WT [cols][rows]
// ---------------------------------------------------------------------------
__global__ __launch_bounds__(256)
void k_transpose_bf16(const float* __restrict__ W, unsigned short* __restrict__ WT,
                      int rows, int cols) {
    __shared__ float tile[32][33];
    int c0 = blockIdx.x * 32, r0 = blockIdx.y * 32;
    int tx = threadIdx.x & 31, ty = threadIdx.x >> 5;   // 8 rows of 32
#pragma unroll
    for (int i = ty; i < 32; i += 8) {
        int r = r0 + i, c = c0 + tx;
        tile[i][tx] = (r < rows && c < cols) ? W[(size_t)r * cols + c] : 0.0f;
    }
    __syncthreads();
#pragma unroll
    for (int i = ty; i < 32; i += 8) {
        int c = c0 + i, r = r0 + tx;
        if (c < cols && r < rows)
            WT[(size_t)c * rows + r] = f32_to_bf16(tile[tx][i]);
    }
}

// ---------------------------------------------------------------------------
// Kernel 2: QKV projection  Y = X[8192,128] @ W[128,2048] + b
//   MODE 0: store [B,H,S,D] (Q, K);  MODE 1: store [B,H,D,S] (V transposed)
//   block = 256 threads = 8 waves, tile M=128 N=64, wave tile 32x32, K=128
// ---------------------------------------------------------------------------
template <int MODE>
__global__ __launch_bounds__(256)
void k_qkv_proj(const float* __restrict__ X, const unsigned short* __restrict__ WT,
                const float* __restrict__ bias, unsigned short* __restrict__ out) {
    __shared__ __align__(16) unsigned short Xs[128][32];
    __shared__ __align__(16) unsigned short Ws[64][32];
    const int m0 = blockIdx.y * 128, n0 = blockIdx.x * 64;
    const int t = threadIdx.x, lane = t & 31, wave = t >> 5;
    const int wm = wave >> 1, wn = wave & 1;
    const int hi = lane >> 4, col = lane & 15;

    v8f acc[2][2];
#pragma unroll
    for (int mi = 0; mi < 2; ++mi)
#pragma unroll
        for (int ni = 0; ni < 2; ++ni) acc[mi][ni] = vzero8();

#pragma unroll
    for (int kc = 0; kc < 4; ++kc) {
        // stage X tile (fp32 -> bf16): 128 rows x 32 k
#pragma unroll
        for (int i = 0; i < 4; ++i) {
            int u = t + i * 256;            // 1024 float4s
            int row = u >> 3, c4 = u & 7;
            float4 xv = *(const float4*)(X + (size_t)(m0 + row) * DD + kc * 32 + c4 * 4);
            unsigned short* dst = &Xs[row][c4 * 4];
            dst[0] = f32_to_bf16(xv.x); dst[1] = f32_to_bf16(xv.y);
            dst[2] = f32_to_bf16(xv.z); dst[3] = f32_to_bf16(xv.w);
        }
        // stage W^T tile: 64 n-rows x 32 k (bf16, contiguous)
        {
            int n = t >> 2, seg = t & 3;
            *(uint4*)&Ws[n][seg * 8] =
                *(const uint4*)(WT + (size_t)(n0 + n) * DD + kc * 32 + seg * 8);
        }
        __syncthreads();

        v16bf a[2], b[2];
#pragma unroll
        for (int mi = 0; mi < 2; ++mi) a[mi] = frag_a(&Xs[wm * 32 + mi * 16 + col][0], hi);
#pragma unroll
        for (int ni = 0; ni < 2; ++ni) b[ni] = frag_b(&Ws[wn * 32 + ni * 16 + col][0], hi);
#pragma unroll
        for (int mi = 0; mi < 2; ++mi)
#pragma unroll
            for (int ni = 0; ni < 2; ++ni)
                acc[mi][ni] = wmma_bf16(a[mi], b[ni], acc[mi][ni]);
        __syncthreads();
    }

    // epilogue: bias + scatter into per-head layout
#pragma unroll
    for (int mi = 0; mi < 2; ++mi) {
#pragma unroll
        for (int ni = 0; ni < 2; ++ni) {
            int cg = n0 + wn * 32 + ni * 16 + col;
            float bv = bias[cg];
            int h = cg >> 7, d = cg & 127;
#pragma unroll
            for (int e = 0; e < 8; ++e) {
                int rg = m0 + wm * 32 + mi * 16 + e + 8 * hi;
                int bidx = rg >> 10, s = rg & 1023;
                float v = acc[mi][ni][e] + bv;
                size_t idx;
                if (MODE == 0) idx = ((size_t)(bidx * HH + h) * SS + s) * DD + d;
                else           idx = ((size_t)(bidx * HH + h) * DD + d) * SS + s;
                out[idx] = f32_to_bf16(v);
            }
        }
    }
}

// ---------------------------------------------------------------------------
// Kernel 3: causal flash attention per (b,h).
//   grid.x = S/128 (query block), grid.y = B*H.  256 threads = 8 waves,
//   each wave owns 16 query rows; key tiles of 32; online softmax in f32.
//   K/V tiles double-buffered in LDS via async global->LDS copies:
//   next tile streams in while WMMAs consume the current one.
// ---------------------------------------------------------------------------
__global__ __launch_bounds__(256)
void k_flash_attn(const unsigned short* __restrict__ Q,
                  const unsigned short* __restrict__ K,
                  const unsigned short* __restrict__ Vt,
                  unsigned short* __restrict__ Ctx) {
    __shared__ __align__(16) unsigned short Qs[128 * 128];
    __shared__ __align__(16) unsigned short Ks[2][32 * 128];
    __shared__ __align__(16) unsigned short Vts[2][128 * 32];
    __shared__ __align__(16) unsigned short Ps[8][16 * 32];

    const int t = threadIdx.x, lane = t & 31, w = t >> 5;
    const int hi = lane >> 4, col = lane & 15;
    const int bh = blockIdx.y;
    const int q0 = blockIdx.x * 128;

    const unsigned short* Qbase = Q + ((size_t)bh * SS + q0) * DD;
    const unsigned short* Kbase = K + (size_t)bh * SS * DD;
    const unsigned short* Vbase = Vt + (size_t)bh * DD * SS;

    const int nkt = blockIdx.x * 4 + 4;         // keys 0 .. q0+127

    // stage the 128x128 query block (contiguous 32KB) asynchronously
#pragma unroll
    for (int i = 0; i < 8; ++i)
        cp_async_b128(&((uint4*)Qs)[t + i * 256], &((const uint4*)Qbase)[t + i * 256]);

    // stage first K/V tile into buffer 0
    {
        const unsigned short* kb = Kbase;
#pragma unroll
        for (int i = 0; i < 2; ++i)
            cp_async_b128(&((uint4*)Ks[0])[t + i * 256], &((const uint4*)kb)[t + i * 256]);
#pragma unroll
        for (int i = 0; i < 2; ++i) {
            int u = t + i * 256; int d = u >> 2, seg = u & 3;
            cp_async_b128(&Vts[0][d * 32 + seg * 8], Vbase + (size_t)d * SS + seg * 8);
        }
    }

    v8f acc[8];
#pragma unroll
    for (int od = 0; od < 8; ++od) acc[od] = vzero8();
    float m[8], l[8];
#pragma unroll
    for (int e = 0; e < 8; ++e) { m[e] = -1e30f; l[e] = 0.0f; }

    const float scale = 0.08838834764831845f;   // 1/sqrt(128)
    const float L2E = 1.4426950408889634f;

    for (int kt = 0; kt < nkt; ++kt) {
        async_wait0();          // current buffer (and Q on first iter) resident
        __syncthreads();        // visible to all waves; prior compute done
        const int cur = kt & 1;

        // stream next K/V tile into the other buffer while computing
        if (kt + 1 < nkt) {
            const int nxt = cur ^ 1;
            const unsigned short* kb = Kbase + (size_t)((kt + 1) * 32) * DD;
#pragma unroll
            for (int i = 0; i < 2; ++i)
                cp_async_b128(&((uint4*)Ks[nxt])[t + i * 256],
                              &((const uint4*)kb)[t + i * 256]);
#pragma unroll
            for (int i = 0; i < 2; ++i) {
                int u = t + i * 256; int d = u >> 2, seg = u & 3;
                cp_async_b128(&Vts[nxt][d * 32 + seg * 8],
                              Vbase + (size_t)d * SS + (kt + 1) * 32 + seg * 8);
            }
        }

        // scores: S[16q x 32k] = Q(16x128) . K^T, K-dim in 4 chunks of 32
        v8f s0 = vzero8(), s1 = vzero8();
#pragma unroll
        for (int dc = 0; dc < 4; ++dc) {
            v16bf aq  = frag_a(&Qs[(w * 16 + col) * DD + dc * 32], hi);
            v16bf bk0 = frag_b(&Ks[cur][(col) * DD + dc * 32], hi);
            v16bf bk1 = frag_b(&Ks[cur][(16 + col) * DD + dc * 32], hi);
            s0 = wmma_bf16(aq, bk0, s0);
            s1 = wmma_bf16(aq, bk1, s1);
        }

        // online softmax (C layout: lane holds rows e+8*hi, col = lane&15)
        unsigned short* Psw = Ps[w];
        const int ktb = kt * 32;
#pragma unroll
        for (int e = 0; e < 8; ++e) {
            int r = e + 8 * hi;
            int qg = q0 + w * 16 + r;
            float x0 = s0[e] * scale;
            float x1 = s1[e] * scale;
            if (ktb + col > qg)      x0 = -1e30f;
            if (ktb + 16 + col > qg) x1 = -1e30f;
            float mx = fmaxf(x0, x1);
#pragma unroll
            for (int off = 8; off > 0; off >>= 1)
                mx = fmaxf(mx, __shfl_xor(mx, off, 32));
            float mnew = fmaxf(m[e], mx);
            float al = exp2f((m[e] - mnew) * L2E);
            float p0 = exp2f((x0 - mnew) * L2E);
            float p1 = exp2f((x1 - mnew) * L2E);
            float rs = p0 + p1;
#pragma unroll
            for (int off = 8; off > 0; off >>= 1)
                rs += __shfl_xor(rs, off, 32);
            l[e] = l[e] * al + rs;
            m[e] = mnew;
#pragma unroll
            for (int od = 0; od < 8; ++od) acc[od][e] *= al;
            Psw[r * 32 + col]      = f32_to_bf16(p0);
            Psw[r * 32 + 16 + col] = f32_to_bf16(p1);
        }

        // ctx += P(16x32) . V(32x128): re-layout P via per-wave LDS scratch
        v16bf pf = frag_a(&Psw[col * 32], hi);
#pragma unroll
        for (int od = 0; od < 8; ++od) {
            v16bf vf = frag_b(&Vts[cur][(od * 16 + col) * 32], hi);
            acc[od] = wmma_bf16(pf, vf, acc[od]);
        }
        // no trailing barrier: next tile lands in the other buffer
    }

    // epilogue: normalize and store ctx bf16 [B*S, H*D]
    const int b = bh >> 4, h = bh & 15;
#pragma unroll
    for (int e = 0; e < 8; ++e) {
        float rcp = 1.0f / l[e];
        int r = e + 8 * hi;
        size_t row = (size_t)b * SS + (q0 + w * 16 + r);
#pragma unroll
        for (int od = 0; od < 8; ++od) {
            int c = h * DD + od * 16 + col;
            Ctx[row * HD + c] = f32_to_bf16(acc[od][e] * rcp);
        }
    }
}

// ---------------------------------------------------------------------------
// Kernel 4: output projection  out = Ctx[8192,2048] @ Wo[2048,2048] + bo (f32)
//   block = 256 threads = 8 waves, tile M=128 N=128, wave tile 32x64 (2x4),
//   K=2048 in 64 chunks, async double-buffered LDS staging.
// ---------------------------------------------------------------------------
__global__ __launch_bounds__(256)
void k_out_proj(const unsigned short* __restrict__ A,
                const unsigned short* __restrict__ WT,
                const float* __restrict__ bias, float* __restrict__ out) {
    __shared__ __align__(16) unsigned short As[2][128][32];
    __shared__ __align__(16) unsigned short Ws[2][128][32];
    const int m0 = blockIdx.y * 128, n0 = blockIdx.x * 128;
    const int t = threadIdx.x, lane = t & 31, wave = t >> 5;
    const int wm = wave >> 1, wn = wave & 1;      // 4 x 2 wave grid
    const int hi = lane >> 4, col = lane & 15;
    const int srow = t >> 2, sseg = t & 3;        // staging: 64 rows x 4 segs (x2)

    v8f acc[2][4];
#pragma unroll
    for (int mi = 0; mi < 2; ++mi)
#pragma unroll
        for (int ni = 0; ni < 4; ++ni) acc[mi][ni] = vzero8();

    auto stage = [&](int kc, int buf) {
        const size_t ko = (size_t)kc * 32 + sseg * 8;
#pragma unroll
        for (int i = 0; i < 2; ++i) {
            cp_async_b128(&As[buf][srow + i * 64][sseg * 8],
                          A + (size_t)(m0 + srow + i * 64) * HD + ko);
            cp_async_b128(&Ws[buf][srow + i * 64][sseg * 8],
                          WT + (size_t)(n0 + srow + i * 64) * HD + ko);
        }
    };

    stage(0, 0);

    for (int kc = 0; kc < HD / 32; ++kc) {
        async_wait0();
        __syncthreads();
        const int cur = kc & 1;
        if (kc + 1 < HD / 32) stage(kc + 1, cur ^ 1);

        v16bf a[2], b[4];
#pragma unroll
        for (int mi = 0; mi < 2; ++mi)
            a[mi] = frag_a(&As[cur][wm * 32 + mi * 16 + col][0], hi);
#pragma unroll
        for (int ni = 0; ni < 4; ++ni)
            b[ni] = frag_b(&Ws[cur][wn * 64 + ni * 16 + col][0], hi);
#pragma unroll
        for (int mi = 0; mi < 2; ++mi)
#pragma unroll
            for (int ni = 0; ni < 4; ++ni)
                acc[mi][ni] = wmma_bf16(a[mi], b[ni], acc[mi][ni]);
        // no trailing barrier: next tile lands in the other buffer
    }

#pragma unroll
    for (int mi = 0; mi < 2; ++mi) {
#pragma unroll
        for (int ni = 0; ni < 4; ++ni) {
            int cg = n0 + wn * 64 + ni * 16 + col;
            float bv = bias[cg];
#pragma unroll
            for (int e = 0; e < 8; ++e) {
                int rg = m0 + wm * 32 + mi * 16 + e + 8 * hi;
                out[(size_t)rg * EE + cg] = acc[mi][ni][e] + bv;
            }
        }
    }
}

// ---------------------------------------------------------------------------
// Host launcher
// ---------------------------------------------------------------------------
extern "C" void kernel_launch(void* const* d_in, const int* in_sizes, int n_in,
                              void* d_out, int out_size, void* d_ws, size_t ws_size,
                              hipStream_t stream) {
    const float* q_matrix = (const float*)d_in[0];
    const float* k_matrix = (const float*)d_in[1];
    const float* v_matrix = (const float*)d_in[2];
    const float* Wq = (const float*)d_in[3];
    const float* bq = (const float*)d_in[4];
    const float* Wk = (const float*)d_in[5];
    const float* bk = (const float*)d_in[6];
    const float* Wv = (const float*)d_in[7];
    const float* bv = (const float*)d_in[8];
    const float* Wo = (const float*)d_in[9];
    const float* bo = (const float*)d_in[10];
    float* out = (float*)d_out;

    char* ws = (char*)d_ws;
    size_t off = 0;
    unsigned short* WqT = (unsigned short*)(ws + off); off += (size_t)HD * DD * 2;
    unsigned short* WkT = (unsigned short*)(ws + off); off += (size_t)HD * DD * 2;
    unsigned short* WvT = (unsigned short*)(ws + off); off += (size_t)HD * DD * 2;
    unsigned short* WoT = (unsigned short*)(ws + off); off += (size_t)EE * HD * 2;
    unsigned short* Qbf = (unsigned short*)(ws + off); off += (size_t)BB * HH * SS * DD * 2;
    unsigned short* Kbf = (unsigned short*)(ws + off); off += (size_t)BB * HH * SS * DD * 2;
    unsigned short* VbT = (unsigned short*)(ws + off); off += (size_t)BB * HH * DD * SS * 2;
    unsigned short* Ctx = (unsigned short*)(ws + off); off += (size_t)BB * SS * HD * 2;

    // 1) weight transposes (fp32 -> bf16, [K,N] -> [N,K])
    k_transpose_bf16<<<dim3(HD / 32, DD / 32), 256, 0, stream>>>(Wq, WqT, DD, HD);
    k_transpose_bf16<<<dim3(HD / 32, DD / 32), 256, 0, stream>>>(Wk, WkT, DD, HD);
    k_transpose_bf16<<<dim3(HD / 32, DD / 32), 256, 0, stream>>>(Wv, WvT, DD, HD);
    k_transpose_bf16<<<dim3(EE / 32, HD / 32), 256, 0, stream>>>(Wo, WoT, HD, EE);

    // 2) QKV projections (V stored transposed per head)
    dim3 gProj(HD / 64, (BB * SS) / 128);
    k_qkv_proj<0><<<gProj, 256, 0, stream>>>(q_matrix, WqT, bq, Qbf);
    k_qkv_proj<0><<<gProj, 256, 0, stream>>>(k_matrix, WkT, bk, Kbf);
    k_qkv_proj<1><<<gProj, 256, 0, stream>>>(v_matrix, WvT, bv, VbT);

    // 3) causal flash attention (128-query blocks, 8 waves, async K/V pipe)
    k_flash_attn<<<dim3(SS / 128, BB * HH), 256, 0, stream>>>(Qbf, Kbf, VbT, Ctx);

    // 4) output projection (128x128 block tile, async double-buffered)
    k_out_proj<<<dim3(EE / 128, (BB * SS) / 128), 256, 0, stream>>>(Ctx, WoT, bo, out);
}
